// PhaseSynchronizedAttention_2113123910370
// MI455X (gfx1250) — matrix operand order repaired and checked by reading, not compiled, not run on hardware
//
#include <hip/hip_runtime.h>
#include <hip/hip_bf16.h>

typedef __attribute__((ext_vector_type(16))) _Float16 v16h;
typedef __attribute__((ext_vector_type(8)))  _Float16 v8h;
typedef __attribute__((ext_vector_type(8)))  float    v8f;
typedef __attribute__((ext_vector_type(4)))  unsigned int u32x4;
typedef __attribute__((ext_vector_type(8)))  int          i32x8;
typedef __attribute__((ext_vector_type(4)))  int          i32x4;

#define D_MODEL 1024
#define N_HEADS 16
#define D_KH    64
#define SEQ     2048
#define BATCH   2
#define M_TOTAL (BATCH * SEQ)   // 4096 rows
#define SROW    2056            // padded LDS score row stride (halves), keeps 16B alignment

// ---------- WMMA helpers ----------

__device__ __forceinline__ v16h frag_cat(v8h lo, v8h hi) {
  return __builtin_shufflevector(lo, hi, 0,1,2,3,4,5,6,7,8,9,10,11,12,13,14,15);
}

// A fragment (16x32, f16): rows m0..m0+15, K = kk..kk+31 of row-major [*, ld].
// Lanes 0-15 hold M=lane, K {0..7, 16..23}; lanes 16-31 hold K {8..15, 24..31}.
__device__ __forceinline__ v16h load_A(const _Float16* base, int ld, int m0, int kk, int lane) {
  const _Float16* p = base + (size_t)(m0 + (lane & 15)) * ld + kk + ((lane >> 4) << 3);
  v8h lo = *(const v8h*)(p);
  v8h hi = *(const v8h*)(p + 16);
  return frag_cat(lo, hi);
}

// B fragment (32x16, f16) where B[k][n] = R[n][k] (R row-major along k).
// Lane holds column n = lane&15, K {0..7, 16..23} (lanes 0-15) / {8..15, 24..31} (lanes 16-31).
__device__ __forceinline__ v16h load_B(const _Float16* R, int ld, int n0, int kk, int lane) {
  const _Float16* p = R + (size_t)(n0 + (lane & 15)) * ld + kk + ((lane >> 4) << 3);
  v8h lo = *(const v8h*)(p);
  v8h hi = *(const v8h*)(p + 16);
  return frag_cat(lo, hi);
}

__device__ __forceinline__ v8f wmma16(v16h a, v16h b, v8f c) {
  return __builtin_amdgcn_wmma_f32_16x16x32_f16(false, a, false, b, (short)0, c, false, false);
}

__device__ __forceinline__ v8f vzero8() { v8f z = {0.f,0.f,0.f,0.f,0.f,0.f,0.f,0.f}; return z; }

// ---------- f32 -> f16 conversion (8 elems/thread: 2x b128 load -> 1x b128 store) ----------

__global__ void cvt_kernel(const float4* __restrict__ in, v8h* __restrict__ out, int n8) {
  int i = blockIdx.x * blockDim.x + threadIdx.x;
  if (i < n8) {
    float4 a = in[i * 2], b = in[i * 2 + 1];
    v8h o;
    o[0] = (_Float16)a.x; o[1] = (_Float16)a.y; o[2] = (_Float16)a.z; o[3] = (_Float16)a.w;
    o[4] = (_Float16)b.x; o[5] = (_Float16)b.y; o[6] = (_Float16)b.z; o[7] = (_Float16)b.w;
    out[i] = o;
  }
}

// ---------- Projection GEMM: out = X @ W^T + bias, optional phase rotation ----------
// Each wave computes a 32(M) x 64(N) tile: 8 accumulators, 2 A-frags share 4 B-frags.
// Software-pipelined: next k-step's fragments are in flight during current WMMAs.
// mode 0/1 : Q/K -> rotate, write head-split row-major [b][h][s][64] to out16
// mode 2   : V    -> write transposed  [b][h][64][s]                 to out16
// mode 3   : O    -> write f32 [4096][1024]                          to out32
__global__ __launch_bounds__(256) void proj_kernel(
    const _Float16* __restrict__ X, const _Float16* __restrict__ W,
    const float* __restrict__ bias, const float* __restrict__ phase,
    _Float16* __restrict__ out16, float* __restrict__ out32, int mode)
{
  const int lane = threadIdx.x & 31;
  const int wid  = blockIdx.x * 8 + (threadIdx.x >> 5);  // 2048 waves total
  const int mt   = wid >> 4;        // 0..127  (32-row tile)
  const int ng   = wid & 15;        // 0..15   (64-col group == head for QKV)
  const int m0   = mt * 32;
  const int nb   = ng * 64;

  v8f acc[2][4];
  #pragma unroll
  for (int h2 = 0; h2 < 2; ++h2)
    #pragma unroll
    for (int t = 0; t < 4; ++t) acc[h2][t] = vzero8();

  // prologue fragments
  v16h a0 = load_A(X, D_MODEL, m0,      0, lane);
  v16h a1 = load_A(X, D_MODEL, m0 + 16, 0, lane);
  v16h b0 = load_B(W, D_MODEL, nb,      0, lane);
  v16h b1 = load_B(W, D_MODEL, nb + 16, 0, lane);
  v16h b2 = load_B(W, D_MODEL, nb + 32, 0, lane);
  v16h b3 = load_B(W, D_MODEL, nb + 48, 0, lane);

  for (int kk = 0; kk < D_MODEL - 32; kk += 32) {
    const int kn = kk + 32;
    // prefetch next k-step while current WMMAs run
    v16h na0 = load_A(X, D_MODEL, m0,      kn, lane);
    v16h na1 = load_A(X, D_MODEL, m0 + 16, kn, lane);
    v16h nb0 = load_B(W, D_MODEL, nb,      kn, lane);
    v16h nb1 = load_B(W, D_MODEL, nb + 16, kn, lane);
    v16h nb2 = load_B(W, D_MODEL, nb + 32, kn, lane);
    v16h nb3 = load_B(W, D_MODEL, nb + 48, kn, lane);
    acc[0][0] = wmma16(a0, b0, acc[0][0]);
    acc[1][0] = wmma16(a1, b0, acc[1][0]);
    acc[0][1] = wmma16(a0, b1, acc[0][1]);
    acc[1][1] = wmma16(a1, b1, acc[1][1]);
    acc[0][2] = wmma16(a0, b2, acc[0][2]);
    acc[1][2] = wmma16(a1, b2, acc[1][2]);
    acc[0][3] = wmma16(a0, b3, acc[0][3]);
    acc[1][3] = wmma16(a1, b3, acc[1][3]);
    a0 = na0; a1 = na1; b0 = nb0; b1 = nb1; b2 = nb2; b3 = nb3;
  }
  // epilogue k-step
  acc[0][0] = wmma16(a0, b0, acc[0][0]);
  acc[1][0] = wmma16(a1, b0, acc[1][0]);
  acc[0][1] = wmma16(a0, b1, acc[0][1]);
  acc[1][1] = wmma16(a1, b1, acc[1][1]);
  acc[0][2] = wmma16(a0, b2, acc[0][2]);
  acc[1][2] = wmma16(a1, b2, acc[1][2]);
  acc[0][3] = wmma16(a0, b3, acc[0][3]);
  acc[1][3] = wmma16(a1, b3, acc[1][3]);

  // bias (per output column)
  #pragma unroll
  for (int t = 0; t < 4; ++t) {
    float bs = bias[nb + t * 16 + (lane & 15)];
    #pragma unroll
    for (int h2 = 0; h2 < 2; ++h2)
      #pragma unroll
      for (int r = 0; r < 8; ++r) acc[h2][t][r] += bs;
  }

  const int bb    = m0 >> 11;        // batch index (2048 rows/batch; 32-row tile never crosses)
  const int srow0 = m0 & (SEQ - 1);  // seq position of tile row 0

  if (mode <= 1) {
    // phase rotation within head: d<32 pairs with d+32 -> acc[.][t] with acc[.][t+2]
    float ph = phase[ng];
    float c = __cosf(ph), s = __sinf(ph);
    #pragma unroll
    for (int h2 = 0; h2 < 2; ++h2)
      #pragma unroll
      for (int t = 0; t < 2; ++t) {
        v8f xr = acc[h2][t], xi = acc[h2][t + 2];
        acc[h2][t]     = xr * c - xi * s;
        acc[h2][t + 2] = xr * s + xi * c;
      }
    _Float16* O = out16 + ((size_t)(bb * N_HEADS + ng) * SEQ) * D_KH;
    #pragma unroll
    for (int h2 = 0; h2 < 2; ++h2)
      #pragma unroll
      for (int t = 0; t < 4; ++t) {
        int d = t * 16 + (lane & 15);
        #pragma unroll
        for (int r = 0; r < 8; ++r) {
          int sr = srow0 + h2 * 16 + r + ((lane >> 4) << 3);
          O[(size_t)sr * D_KH + d] = (_Float16)acc[h2][t][r];
        }
      }
  } else if (mode == 2) {
    // transposed store: lane owns column d, 8 consecutive seq positions -> one 16B store
    #pragma unroll
    for (int h2 = 0; h2 < 2; ++h2) {
      int s0 = srow0 + h2 * 16 + ((lane >> 4) << 3);
      #pragma unroll
      for (int t = 0; t < 4; ++t) {
        int d = t * 16 + (lane & 15);
        _Float16* p = out16 + ((size_t)(bb * N_HEADS + ng) * D_KH + d) * SEQ + s0;
        v8h pk;
        #pragma unroll
        for (int r = 0; r < 8; ++r) pk[r] = (_Float16)acc[h2][t][r];
        *(v8h*)p = pk;
      }
    }
  } else {
    #pragma unroll
    for (int h2 = 0; h2 < 2; ++h2)
      #pragma unroll
      for (int t = 0; t < 4; ++t) {
        int n = nb + t * 16 + (lane & 15);
        #pragma unroll
        for (int r = 0; r < 8; ++r) {
          int m = m0 + h2 * 16 + r + ((lane >> 4) << 3);
          out32[(size_t)m * D_MODEL + n] = acc[h2][t][r];
        }
      }
  }
}

// ---------- Attention: one workgroup (4 waves) per (b, h, 16-query tile) ----------
// Q tile staged into LDS by the Tensor Data Mover (one TDM descriptor from wave 0).
__global__ __launch_bounds__(128) void attn_kernel(
    const _Float16* __restrict__ Qh, const _Float16* __restrict__ Kh,
    const _Float16* __restrict__ Vt, _Float16* __restrict__ O16)
{
  __shared__ _Float16 qtile[16 * D_KH];                   // 2 KB static LDS for the Q tile
  extern __shared__ char smem[];
  _Float16* sc   = (_Float16*)smem;                       // [16][SROW] f16 scores -> probs
  float*    red  = (float*)(smem + 16 * SROW * 2);        // [16][8] partials
  float*    rmax = red + 128;                             // [16]
  float*    rinv = rmax + 16;                             // [16]

  const int tid  = threadIdx.x;
  const int lane = tid & 31;
  const int w    = tid >> 5;   // wave 0..3
  const int qt   = blockIdx.x, h = blockIdx.y, b = blockIdx.z;
  const int bh   = b * N_HEADS + h;
  const _Float16* Q = Qh + (size_t)bh * SEQ * D_KH;
  const _Float16* K = Kh + (size_t)bh * SEQ * D_KH;
  const _Float16* V = Vt + (size_t)bh * D_KH * SEQ;
  const int m0 = qt * 16;

  // ---- TDM: stage 16x64 f16 Q tile into LDS (wave 0 issues one descriptor) ----
  if (tid < 32) {
    unsigned long long ga = (unsigned long long)(uintptr_t)(Q + (size_t)m0 * D_KH);
    unsigned int lds_off  = (unsigned int)(uintptr_t)qtile;  // LDS generic addr low 32 = offset
    u32x4 g0;
    g0[0] = 1u;                                            // count=1, user descriptor
    g0[1] = lds_off;                                       // LDS destination (bytes)
    g0[2] = (unsigned int)ga;                              // global_addr[31:0]
    g0[3] = (unsigned int)((ga >> 32) & 0x01FFFFFFu) | 0x80000000u; // addr[56:32] | type=2
    i32x8 g1;
    g1[0] = 0x00010000;                                    // data_size=1 (2B), no multicast
    g1[1] = (int)(64u << 16);                              // tensor_dim0 = 64 (bits 79:48 lo)
    g1[2] = (int)(2048u << 16);                            // tensor_dim1 = 2048 (bits 111:80 lo)
    g1[3] = (int)(64u << 16);                              // tile_dim0 = 64 (bits 127:112)
    g1[4] = 16;                                            // tile_dim1 = 16 (bits 143:128)
    g1[5] = 64;                                            // tensor_dim0_stride = 64
    g1[6] = 0;
    g1[7] = 0;
    i32x4 z4; z4[0] = 0; z4[1] = 0; z4[2] = 0; z4[3] = 0;  // groups 2/3 unused (2-D tensor)
    i32x8 z8;
    #pragma unroll
    for (int i = 0; i < 8; ++i) z8[i] = 0;
    __builtin_amdgcn_tensor_load_to_lds(g0, g1, z4, z4, z8, 0);
    __builtin_amdgcn_s_wait_tensorcnt(0);
  }
  __syncthreads();

  // ---- Phase A: S = (Q K^T) / sqrt(d) for all 2048 keys (pipelined K frags) ----
  v16h aq0 = load_A(qtile, D_KH, 0, 0,  lane);
  v16h aq1 = load_A(qtile, D_KH, 0, 32, lane);
  v16h bk0 = load_B(K, D_KH, w * 16, 0,  lane);
  v16h bk1 = load_B(K, D_KH, w * 16, 32, lane);
  for (int nt = w; nt < SEQ / 16; nt += 4) {
    int ntn = nt + 4;
    v16h nb0 = bk0, nb1 = bk1;
    if (ntn < SEQ / 16) {
      nb0 = load_B(K, D_KH, ntn * 16, 0,  lane);
      nb1 = load_B(K, D_KH, ntn * 16, 32, lane);
    }
    v8f acc = vzero8();
    acc = wmma16(aq0, bk0, acc);
    acc = wmma16(aq1, bk1, acc);
    int col = nt * 16 + (lane & 15);
    #pragma unroll
    for (int r = 0; r < 8; ++r) {
      int row = r + ((lane >> 4) << 3);
      sc[row * SROW + col] = (_Float16)(acc[r] * 0.125f);  // 1/sqrt(64)
    }
    bk0 = nb0; bk1 = nb1;
  }
  __syncthreads();

  // ---- Phase B: softmax over 2048 keys per row (f32 math over f16 LDS) ----
  const int row = tid & 15, chunk = tid >> 4;  // 8 threads per row, 256 cols each
  const int c0 = chunk * 256;
  float mx = -1e30f;
  for (int j = 0; j < 256; ++j)
    mx = fmaxf(mx, (float)sc[row * SROW + c0 + j]);
  red[row * 8 + chunk] = mx;
  __syncthreads();
  if (tid < 16) {
    float m2 = red[tid * 8];
    #pragma unroll
    for (int g = 1; g < 8; ++g) m2 = fmaxf(m2, red[tid * 8 + g]);
    rmax[tid] = m2;
  }
  __syncthreads();
  const float rm = rmax[row];
  float sum = 0.f;
  for (int j = 0; j < 256; ++j)
    sum += __expf((float)sc[row * SROW + c0 + j] - rm);
  red[row * 8 + chunk] = sum;
  __syncthreads();
  if (tid < 16) {
    float s2 = 0.f;
    #pragma unroll
    for (int g = 0; g < 8; ++g) s2 += red[tid * 8 + g];
    rinv[tid] = 1.0f / s2;
  }
  __syncthreads();
  const float ri = rinv[row];
  for (int j = 0; j < 256; ++j) {
    size_t idx = (size_t)row * SROW + c0 + j;
    sc[idx] = (_Float16)(__expf((float)sc[idx] - rm) * ri);
  }
  __syncthreads();

  // ---- Phase C: out = P @ V (wave w owns d-cols w*16..+15; K = 2048, pipelined) ----
  v8f o = vzero8();
  const int n0 = w * 16;
  v16h pa = load_A(sc, SROW, 0, 0, lane);
  v16h pb = load_B(V, SEQ, n0, 0, lane);
  for (int kk = 0; kk < SEQ; kk += 32) {
    int kn = kk + 32;
    v16h na = pa, nbf = pb;
    if (kn < SEQ) {
      na  = load_A(sc, SROW, 0, kn, lane);
      nbf = load_B(V, SEQ, n0, kn, lane);
    }
    o = wmma16(pa, pb, o);
    pa = na; pb = nbf;
  }
  const int dcol = h * D_KH + n0 + (lane & 15);
  #pragma unroll
  for (int r = 0; r < 8; ++r) {
    int sr = m0 + r + ((lane >> 4) << 3);
    O16[((size_t)b * SEQ + sr) * D_MODEL + dcol] = (_Float16)o[r];
  }
}

// ---------- Host launcher ----------

extern "C" void kernel_launch(void* const* d_in, const int* in_sizes, int n_in,
                              void* d_out, int out_size, void* d_ws, size_t ws_size,
                              hipStream_t stream) {
  const float* q     = (const float*)d_in[0];
  const float* k     = (const float*)d_in[1];
  const float* v     = (const float*)d_in[2];
  const float* Wq    = (const float*)d_in[3];
  const float* bq    = (const float*)d_in[4];
  const float* Wk    = (const float*)d_in[5];
  const float* bk    = (const float*)d_in[6];
  const float* Wv    = (const float*)d_in[7];
  const float* bv    = (const float*)d_in[8];
  const float* Wo    = (const float*)d_in[9];
  const float* bo    = (const float*)d_in[10];
  const float* phase = (const float*)d_in[11];

  const size_t NX = (size_t)M_TOTAL * D_MODEL;   // 4,194,304
  const size_t NW = (size_t)D_MODEL * D_MODEL;   // 1,048,576

  char* ws = (char*)d_ws;
  size_t off = 0;
  auto carve = [&](size_t bytes) { char* p = ws + off; off += bytes; return p; };
  _Float16* X16q = (_Float16*)carve(NX * 2);
  _Float16* X16k = (_Float16*)carve(NX * 2);
  _Float16* X16v = (_Float16*)carve(NX * 2);
  _Float16* W16q = (_Float16*)carve(NW * 2);
  _Float16* W16k = (_Float16*)carve(NW * 2);
  _Float16* W16v = (_Float16*)carve(NW * 2);
  _Float16* W16o = (_Float16*)carve(NW * 2);
  _Float16* Qh   = (_Float16*)carve(NX * 2);     // [b][h][s][64]
  _Float16* Kh16 = (_Float16*)carve(NX * 2);     // [b][h][s][64]
  _Float16* VtT  = (_Float16*)carve(NX * 2);     // [b][h][64][s]
  _Float16* O16  = (_Float16*)carve(NX * 2);     // [b][s][1024]

  // 1) f32 -> f16 conversions (8 elems/thread)
  const int nx8 = (int)(NX / 8), nw8 = (int)(NW / 8);
  cvt_kernel<<<(nx8 + 255) / 256, 256, 0, stream>>>((const float4*)q,  (v8h*)X16q, nx8);
  cvt_kernel<<<(nx8 + 255) / 256, 256, 0, stream>>>((const float4*)k,  (v8h*)X16k, nx8);
  cvt_kernel<<<(nx8 + 255) / 256, 256, 0, stream>>>((const float4*)v,  (v8h*)X16v, nx8);
  cvt_kernel<<<(nw8 + 255) / 256, 256, 0, stream>>>((const float4*)Wq, (v8h*)W16q, nw8);
  cvt_kernel<<<(nw8 + 255) / 256, 256, 0, stream>>>((const float4*)Wk, (v8h*)W16k, nw8);
  cvt_kernel<<<(nw8 + 255) / 256, 256, 0, stream>>>((const float4*)Wv, (v8h*)W16v, nw8);
  cvt_kernel<<<(nw8 + 255) / 256, 256, 0, stream>>>((const float4*)Wo, (v8h*)W16o, nw8);

  // 2) QKV projections + bias + phase rotation (V stored transposed); 2048 waves each
  proj_kernel<<<256, 256, 0, stream>>>(X16q, W16q, bq, phase, Qh,   nullptr, 0);
  proj_kernel<<<256, 256, 0, stream>>>(X16k, W16k, bk, phase, Kh16, nullptr, 1);
  proj_kernel<<<256, 256, 0, stream>>>(X16v, W16v, bv, phase, VtT,  nullptr, 2);

  // 3) attention (dynamic LDS: f16 scores [16][SROW] + reduction scratch)
  const size_t smem_bytes = (size_t)16 * SROW * 2 + 128 * 4 + 16 * 4 + 16 * 4;
  attn_kernel<<<dim3(SEQ / 16, N_HEADS, BATCH), 128, smem_bytes, stream>>>(Qh, Kh16, VtT, O16);

  // 4) output projection -> f32 d_out
  proj_kernel<<<256, 256, 0, stream>>>(O16, W16o, bo, phase, nullptr, (float*)d_out, 3);
}
